// RelGAT_13511967113596
// MI455X (gfx1250) — matrix operand order repaired and checked by reading, not compiled, not run on hardware
//
#include <hip/hip_runtime.h>
#include <math.h>

#define NNODES 100000
#define NREL   4
#define NEDGE  200000
#define DIN    256
#define DHID   64
#define NHEAD  8
#define HO     512   // NHEAD * DHID
#define DOUT   64
#define NEG_SLOPE 0.2f

typedef __attribute__((ext_vector_type(16))) _Float16 v16h;
typedef __attribute__((ext_vector_type(8)))  float    v8f;

struct Frag16 { union { v16h h; uint4 u[2]; }; };

// ---------------- utility kernels ----------------

__global__ void k_zero_f4(float* __restrict__ p, int n4) {
    int i = blockIdx.x * blockDim.x + threadIdx.x;
    if (i < n4) ((float4*)p)[i] = make_float4(0.f, 0.f, 0.f, 0.f);
}

__global__ void k_f32_to_f16(const float* __restrict__ s, _Float16* __restrict__ d, int n) {
    int i = blockIdx.x * blockDim.x + threadIdx.x;
    if (i < n) d[i] = (_Float16)s[i];
}

// W: [K, HO] row-major -> Wt: [HO, K] fp16
__global__ void k_wtrans(const float* __restrict__ W, _Float16* __restrict__ Wt, int K) {
    int i = blockIdx.x * blockDim.x + threadIdx.x;
    if (i >= K * HO) return;
    int n = i / K;
    int k = i - n * K;
    Wt[(size_t)n * K + k] = (_Float16)W[(size_t)k * HO + n];
}

// ---------------- WMMA GEMM + fused el/er ----------------
// Block: 8 waves x 32 rows = 256 rows, one head (64 cols).
// B panel (64 x K fp16) staged once in LDS via async copy, shared by all waves.
__global__ void k_gemm_wmma(const _Float16* __restrict__ A,
                            const _Float16* __restrict__ Wt,
                            float* __restrict__ F,
                            const float* __restrict__ al,
                            const float* __restrict__ ar,
                            float* __restrict__ el,
                            float* __restrict__ er,
                            int nrows, int K) {
    __shared__ _Float16 sB[64 * HO];   // up to 64 rows x K=512 halves = 64KB

    const int lane = threadIdx.x & 31;
    const int wave = threadIdx.x >> 5;
    const int head = blockIdx.y;
    const int n0   = head * 64;
    const int tile_m = blockIdx.x * 8 + wave;
    const int row0   = tile_m * 32;

    // ---- cooperative async stage of B panel: Wt rows [n0, n0+64) are contiguous ----
    {
        const char* gsrc = (const char*)(Wt + (size_t)n0 * K);
        const int total16 = (64 * K * 2) >> 4;          // 16B chunks
        for (int i = threadIdx.x; i < total16; i += 256) {
            unsigned lds = (unsigned)(size_t)&sB[0] + (unsigned)(i << 4);
            unsigned long long ga = (unsigned long long)(gsrc + ((size_t)i << 4));
            asm volatile("global_load_async_to_lds_b128 %0, %1, off"
                         :: "v"(lds), "v"(ga) : "memory");
        }
        asm volatile("s_wait_asynccnt 0" ::: "memory");
        __syncthreads();
    }

    if (row0 < nrows) {                         // wave-uniform: EXEC all-ones inside
        v8f c0 = {}, c1 = {}, c2 = {}, c3 = {};
        v8f c4 = {}, c5 = {}, c6 = {}, c7 = {};

        // A fragment addressing: lane -> M = lane&15 ; 16 contiguous K halves, kb=(lane>>4)*16
        const int mA  = lane & 15;
        const int kbA = (lane >> 4) << 4;
        const _Float16* arow0 = A + (size_t)(row0 + mA) * K + kbA;
        const _Float16* arow1 = A + (size_t)(row0 + 16 + mA) * K + kbA;

        // B fragment from LDS: lane -> N = lane&15 ; 16 contiguous K halves, kb=(lane>>4)*16
        const int nB  = lane & 15;
        const int kbB = (lane >> 4) << 4;
        const _Float16* bp0 = sB + (size_t)nB        * K + kbB;
        const _Float16* bp1 = sB + (size_t)(16 + nB) * K + kbB;
        const _Float16* bp2 = sB + (size_t)(32 + nB) * K + kbB;
        const _Float16* bp3 = sB + (size_t)(48 + nB) * K + kbB;

        for (int k0 = 0; k0 < K; k0 += 32) {
            Frag16 a0, a1, b;
            a0.u[0] = *(const uint4*)(arow0 + k0);
            a0.u[1] = *(const uint4*)(arow0 + k0 + 8);
            a1.u[0] = *(const uint4*)(arow1 + k0);
            a1.u[1] = *(const uint4*)(arow1 + k0 + 8);

            b.u[0] = *(const uint4*)(bp0 + k0); b.u[1] = *(const uint4*)(bp0 + k0 + 8);
            c0 = __builtin_amdgcn_wmma_f32_16x16x32_f16(false, a0.h, false, b.h, (short)0, c0, false, false);
            c4 = __builtin_amdgcn_wmma_f32_16x16x32_f16(false, a1.h, false, b.h, (short)0, c4, false, false);

            b.u[0] = *(const uint4*)(bp1 + k0); b.u[1] = *(const uint4*)(bp1 + k0 + 8);
            c1 = __builtin_amdgcn_wmma_f32_16x16x32_f16(false, a0.h, false, b.h, (short)0, c1, false, false);
            c5 = __builtin_amdgcn_wmma_f32_16x16x32_f16(false, a1.h, false, b.h, (short)0, c5, false, false);

            b.u[0] = *(const uint4*)(bp2 + k0); b.u[1] = *(const uint4*)(bp2 + k0 + 8);
            c2 = __builtin_amdgcn_wmma_f32_16x16x32_f16(false, a0.h, false, b.h, (short)0, c2, false, false);
            c6 = __builtin_amdgcn_wmma_f32_16x16x32_f16(false, a1.h, false, b.h, (short)0, c6, false, false);

            b.u[0] = *(const uint4*)(bp3 + k0); b.u[1] = *(const uint4*)(bp3 + k0 + 8);
            c3 = __builtin_amdgcn_wmma_f32_16x16x32_f16(false, a0.h, false, b.h, (short)0, c3, false, false);
            c7 = __builtin_amdgcn_wmma_f32_16x16x32_f16(false, a1.h, false, b.h, (short)0, c7, false, false);
        }

        // D layout: lane&15 -> N col; VGPR v -> M = v + 8*(lane>>4)
        const int mhi  = (lane >> 4) << 3;
        const int ncol = lane & 15;
        float* frow = F + (size_t)row0 * HO + n0 + ncol;
#pragma unroll
        for (int v = 0; v < 8; ++v) {
            size_t ro = (size_t)(v + mhi) * HO;
            frow[ro]      = c0[v];
            frow[ro + 16] = c1[v];
            frow[ro + 32] = c2[v];
            frow[ro + 48] = c3[v];
            size_t r1 = (size_t)(16 + v + mhi) * HO;
            frow[r1]      = c4[v];
            frow[r1 + 16] = c5[v];
            frow[r1 + 32] = c6[v];
            frow[r1 + 48] = c7[v];
        }

        // ---- fused el/er: per-row dot with attention vectors, butterfly-reduced ----
        float alv[4], arv[4];
#pragma unroll
        for (int t = 0; t < 4; ++t) {
            alv[t] = al[n0 + 16 * t + ncol];
            arv[t] = ar[n0 + 16 * t + ncol];
        }
#pragma unroll
        for (int v = 0; v < 8; ++v) {
            float pl0 = c0[v] * alv[0] + c1[v] * alv[1] + c2[v] * alv[2] + c3[v] * alv[3];
            float pr0 = c0[v] * arv[0] + c1[v] * arv[1] + c2[v] * arv[2] + c3[v] * arv[3];
            float pl1 = c4[v] * alv[0] + c5[v] * alv[1] + c6[v] * alv[2] + c7[v] * alv[3];
            float pr1 = c4[v] * arv[0] + c5[v] * arv[1] + c6[v] * arv[2] + c7[v] * arv[3];
#pragma unroll
            for (int msk = 1; msk < 16; msk <<= 1) {
                pl0 += __shfl_xor(pl0, msk, 32);
                pr0 += __shfl_xor(pr0, msk, 32);
                pl1 += __shfl_xor(pl1, msk, 32);
                pr1 += __shfl_xor(pr1, msk, 32);
            }
            if (ncol == 0) {
                int r0 = row0 + v + mhi;
                el[r0 * NHEAD + head] = pl0;
                er[r0 * NHEAD + head] = pr0;
                int r1 = row0 + 16 + v + mhi;
                el[r1 * NHEAD + head] = pl1;
                er[r1 * NHEAD + head] = pr1;
            }
        }
    }
}

// ---------------- softmax / scatter phase ----------------

__global__ void k_init_mden(float* __restrict__ m, float* __restrict__ den) {
    int i = blockIdx.x * blockDim.x + threadIdx.x;
    if (i < NNODES * NHEAD) { m[i] = -INFINITY; den[i] = 0.f; }
}

__device__ inline void atomicMaxF(float* a, float v) {
    if (v >= 0.f) atomicMax((int*)a, __float_as_int(v));
    else          atomicMin((unsigned int*)a, (unsigned int)__float_as_int(v));
}

// pass1: scores + segment max
__global__ void k_edge1(const int* __restrict__ src, const int* __restrict__ dst,
                        const float* __restrict__ el, const float* __restrict__ er,
                        float* __restrict__ esc, float* __restrict__ m) {
    int tid = blockIdx.x * blockDim.x + threadIdx.x;
    if (tid >= NEDGE * NHEAD) return;
    int e = tid >> 3, h = tid & 7;
    int s = src[e], d = dst[e];
    float sc = el[s * NHEAD + h] + er[d * NHEAD + h];
    sc = (sc > 0.f) ? sc : NEG_SLOPE * sc;
    esc[tid] = sc;
    atomicMaxF(&m[d * NHEAD + h], sc);
}

// pass2: exp + segment sum
__global__ void k_edge2(const int* __restrict__ dst, const float* __restrict__ m,
                        float* __restrict__ esc, float* __restrict__ den) {
    int tid = blockIdx.x * blockDim.x + threadIdx.x;
    if (tid >= NEDGE * NHEAD) return;
    int e = tid >> 3, h = tid & 7;
    int d = dst[e];
    float ex = __expf(esc[tid] - m[d * NHEAD + h]);
    esc[tid] = ex;
    atomicAdd(&den[d * NHEAD + h], ex);
}

// pass3: weighted message scatter (thread per edge,head,float4-chunk)
__global__ void k_edge3(const int* __restrict__ src, const int* __restrict__ dst,
                        const float* __restrict__ esc, const float* __restrict__ den,
                        const float* __restrict__ F, float* __restrict__ Or) {
    int tid = blockIdx.x * blockDim.x + threadIdx.x;
    if (tid >= NEDGE * NHEAD * 16) return;
    int q  = tid & 15;
    int eh = tid >> 4;
    int h  = eh & 7;
    int e  = eh >> 3;
    int s = src[e], d = dst[e];
    float alpha = esc[eh] / (den[d * NHEAD + h] + 1e-9f);
    const float4 fv = *(const float4*)(F + (size_t)s * HO + h * DHID + q * 4);
    float* o = Or + (size_t)d * HO + h * DHID + q * 4;
    atomicAdd(o + 0, fv.x * alpha);
    atomicAdd(o + 1, fv.y * alpha);
    atomicAdd(o + 2, fv.z * alpha);
    atomicAdd(o + 3, fv.w * alpha);
}

__global__ void k_eluacc(const float* __restrict__ Or, float* __restrict__ acc, int n4) {
    int i = blockIdx.x * blockDim.x + threadIdx.x;
    if (i >= n4) return;
    float4 v = ((const float4*)Or)[i];
    float4 a = ((float4*)acc)[i];
    a.x += (v.x > 0.f) ? v.x : (__expf(v.x) - 1.f);
    a.y += (v.y > 0.f) ? v.y : (__expf(v.y) - 1.f);
    a.z += (v.z > 0.f) ? v.z : (__expf(v.z) - 1.f);
    a.w += (v.w > 0.f) ? v.w : (__expf(v.w) - 1.f);
    ((float4*)acc)[i] = a;
}

__global__ void k_mean(const float* __restrict__ acc, float* __restrict__ out) {
    int tid = blockIdx.x * blockDim.x + threadIdx.x;
    if (tid >= NNODES * DOUT) return;
    int n = tid / DOUT, o = tid - n * DOUT;
    float s = 0.f;
#pragma unroll
    for (int h = 0; h < NHEAD; ++h) s += acc[(size_t)n * HO + h * DHID + o];
    out[tid] = s * 0.125f;
}

// ---------------- host driver ----------------

extern "C" void kernel_launch(void* const* d_in, const int* in_sizes, int n_in,
                              void* d_out, int out_size, void* d_ws, size_t ws_size,
                              hipStream_t stream) {
    const float* x   = (const float*)d_in[0];
    const int*   src = (const int*)d_in[1];
    const int*   dst = (const int*)d_in[2];
    const float* W[3]  = {(const float*)d_in[3], (const float*)d_in[6], (const float*)d_in[9]};
    const float* al[3] = {(const float*)d_in[4], (const float*)d_in[7], (const float*)d_in[10]};
    const float* ar[3] = {(const float*)d_in[5], (const float*)d_in[8], (const float*)d_in[11]};

    char* ws = (char*)d_ws;
    size_t off = 0;
    auto alloc = [&](size_t bytes) -> char* {
        char* p = ws + off;
        off += (bytes + 255) & ~(size_t)255;
        return p;
    };
    float*    acc = (float*)alloc((size_t)NNODES * HO * 4);
    float*    F   = (float*)alloc((size_t)NNODES * HO * 4);
    float*    Or  = (float*)alloc((size_t)NNODES * HO * 4);
    _Float16* H16 = (_Float16*)alloc((size_t)NNODES * HO * 2);
    _Float16* Wt  = (_Float16*)alloc((size_t)NREL * (DIN + HO + HO) * HO * 2);
    float*    el  = (float*)alloc((size_t)NNODES * NHEAD * 4);
    float*    er  = (float*)alloc((size_t)NNODES * NHEAD * 4);
    float*    mb  = (float*)alloc((size_t)NNODES * NHEAD * 4);
    float*    den = (float*)alloc((size_t)NNODES * NHEAD * 4);
    float*    esc = (float*)alloc((size_t)NEDGE * NHEAD * 4);

    const int Ks[3] = {DIN, HO, HO};
    size_t wtoff[3];
    wtoff[0] = 0;
    wtoff[1] = (size_t)NREL * HO * DIN;
    wtoff[2] = wtoff[1] + (size_t)NREL * HO * HO;

    // weights -> transposed fp16 (once; stays hot in L2)
    for (int l = 0; l < 3; ++l)
        for (int r = 0; r < NREL; ++r) {
            int K = Ks[l], nel = K * HO;
            k_wtrans<<<(nel + 255) / 256, 256, 0, stream>>>(
                W[l] + (size_t)r * K * HO, Wt + wtoff[l] + (size_t)r * HO * K, K);
        }

    const int n4  = NNODES * HO / 4;
    const int nh  = NNODES * NHEAD;
    const int neh = NEDGE * NHEAD;
    const int ne3 = NEDGE * NHEAD * 16;
    const int tiles_m = NNODES / 32;                       // 3125 (N % 32 == 0)
    const dim3 gemm_grid((tiles_m + 7) / 8, NHEAD);

    for (int l = 0; l < 3; ++l) {
        int K = Ks[l];
        int nelem = NNODES * K;
        // activations -> fp16 (layer 0 reads x; later layers read previous acc)
        k_f32_to_f16<<<(nelem + 255) / 256, 256, 0, stream>>>(l == 0 ? x : acc, H16, nelem);
        k_zero_f4<<<(n4 + 255) / 256, 256, 0, stream>>>(acc, n4);

        for (int r = 0; r < NREL; ++r) {
            const _Float16* Wtr = Wt + wtoff[l] + (size_t)r * HO * K;
            const float* alr = al[l] + (size_t)r * NHEAD * DHID;
            const float* arr = ar[l] + (size_t)r * NHEAD * DHID;

            k_gemm_wmma<<<gemm_grid, 256, 0, stream>>>(H16, Wtr, F, alr, arr, el, er,
                                                       NNODES, K);

            k_init_mden<<<(nh + 255) / 256, 256, 0, stream>>>(mb, den);
            k_zero_f4<<<(n4 + 255) / 256, 256, 0, stream>>>(Or, n4);

            const int* sr = src + (size_t)r * NEDGE;
            const int* dr = dst + (size_t)r * NEDGE;
            k_edge1<<<(neh + 255) / 256, 256, 0, stream>>>(sr, dr, el, er, esc, mb);
            k_edge2<<<(neh + 255) / 256, 256, 0, stream>>>(dr, mb, esc, den);
            k_edge3<<<(ne3 + 255) / 256, 256, 0, stream>>>(sr, dr, esc, den, F, Or);
            k_eluacc<<<(n4 + 255) / 256, 256, 0, stream>>>(Or, acc, n4);
        }
    }

    int no = NNODES * DOUT;
    k_mean<<<(no + 255) / 256, 256, 0, stream>>>(acc, (float*)d_out);
}
